// GATLayer_39178691674903
// MI455X (gfx1250) — matrix operand order repaired and checked by reading, not compiled, not run on hardware
//
#include <hip/hip_runtime.h>

typedef __attribute__((ext_vector_type(16))) _Float16 v16h;
typedef __attribute__((ext_vector_type(8)))  float    v8f;

static constexpr int Nn = 8192;
static constexpr int Dd = 128;
static constexpr int Hh = 4;
static constexpr float LOG2E = 1.4426950408889634f;

// workspace layout (bytes)
static constexpr size_t OFF_XT   = 0;                          // 8192*128 f16 (x transposed)  2 MB
static constexpr size_t OFF_U    = OFF_XT   + (size_t)Nn*Dd*2; // 8*128 f32
static constexpr size_t OFF_C    = OFF_U    + 8*Dd*4;          // 8 f32
static constexpr size_t OFF_S    = OFF_C    + 256;             // 4*8192 f32 (s*log2e, [h][i])
static constexpr size_t OFF_T    = OFF_S    + (size_t)Hh*Nn*4; // 4*8192 f32 (t*log2e, [h][i])
static constexpr size_t OFF_SMAX = OFF_T    + (size_t)Hh*Nn*4; // 4 f32
static constexpr size_t OFF_CC   = OFF_SMAX + 256;             // 4*8192 f32 (-m2 - log2(H*Z))

__device__ __forceinline__ float lrelu01(float v) { return v > 0.f ? v : 0.01f * v; }
__device__ __forceinline__ float fexp2(float v)   { return __builtin_amdgcn_exp2f(v); }
__device__ __forceinline__ float flog2(float v)   { return __builtin_amdgcn_logf(v); }

// ---- Stage 1: f16 transposed copy of x: xT[c][i] = (f16)x[i][c] ----
__global__ void k_castT(const float* __restrict__ x, _Float16* __restrict__ xT) {
    int e = blockIdx.x * 256 + threadIdx.x;     // e over N*D
    int i = e >> 7, c = e & 127;
    xT[(size_t)c * Nn + i] = (_Float16)x[e];
}

// ---- Stage 2: u[idx][k] = sum_m W[m][k]*a[idx part][m], cvec[idx] = b . a_part ----
__global__ void k_prep(const float* __restrict__ W, const float* __restrict__ b,
                       const float* __restrict__ a,
                       float* __restrict__ u, float* __restrict__ cvec) {
    int tid = threadIdx.x;
    for (int q = 0; q < 4; ++q) {
        int flat = q * 256 + tid;               // [0,1024): idx = flat/128, k = flat%128
        int idx = flat >> 7, k = flat & 127;
        const float* av = a + (idx >> 1) * (2 * Dd) + (idx & 1) * Dd;
        float acc = 0.f;
        for (int m = 0; m < Dd; ++m) acc += W[m * Dd + k] * av[m];
        u[flat] = acc;
    }
    if (tid < 8) {
        const float* av = a + (tid >> 1) * (2 * Dd) + (tid & 1) * Dd;
        float acc = 0.f;
        for (int m = 0; m < Dd; ++m) acc += b[m] * av[m];
        cvec[tid] = acc;
    }
}

// ---- Stage 3: s2[h][i] = (x[i].u[2h] + c)*log2e, t2 likewise (log2 domain) ----
__global__ void k_st(const float* __restrict__ x, const float* __restrict__ u,
                     const float* __restrict__ cvec,
                     float* __restrict__ s, float* __restrict__ t) {
    __shared__ float su[8 * Dd];
    __shared__ float sc[8];
    int tid = threadIdx.x;
    for (int q = tid; q < 8 * Dd; q += 256) su[q] = u[q];
    if (tid < 8) sc[tid] = cvec[tid];
    __syncthreads();
    int i = blockIdx.x * 256 + tid;
    float acc[8] = {0.f, 0.f, 0.f, 0.f, 0.f, 0.f, 0.f, 0.f};
    for (int k = 0; k < Dd; ++k) {
        float xv = x[(size_t)i * Dd + k];
        #pragma unroll
        for (int idx = 0; idx < 8; ++idx) acc[idx] += xv * su[idx * Dd + k];
    }
    #pragma unroll
    for (int h = 0; h < Hh; ++h) {
        s[h * Nn + i] = (acc[2 * h]     + sc[2 * h])     * LOG2E;
        t[h * Nn + i] = (acc[2 * h + 1] + sc[2 * h + 1]) * LOG2E;
    }
}

// ---- Stage 4: smax[h] = max_j s2[h][j] (positive scaling preserves argmax) ----
__global__ void k_smax(const float* __restrict__ s, float* __restrict__ smax) {
    __shared__ float red[256];
    int tid = threadIdx.x;
    for (int h = 0; h < Hh; ++h) {
        float mx = -3.4e38f;
        for (int j = tid; j < Nn; j += 256) mx = fmaxf(mx, s[h * Nn + j]);
        red[tid] = mx;
        __syncthreads();
        for (int st = 128; st > 0; st >>= 1) {
            if (tid < st) red[tid] = fmaxf(red[tid], red[tid + st]);
            __syncthreads();
        }
        if (tid == 0) smax[h] = red[0];
        __syncthreads();
    }
}

// ---- Stage 5: C[h][i] = -m2 - log2(H * Z),  Z = sum_j exp2(lrelu(t2+s2_j) - m2) ----
__global__ void k_zc(const float* __restrict__ s, const float* __restrict__ t,
                     const float* __restrict__ smax, float* __restrict__ Cout) {
    int g = blockIdx.x * 256 + threadIdx.x;     // g over H*N
    int h = g >> 13, i = g & (Nn - 1);
    float tv = t[h * Nn + i];
    float m2 = lrelu01(tv + smax[h]);
    const float* sh = s + h * Nn;
    float z0 = 0.f, z1 = 0.f, z2 = 0.f, z3 = 0.f;
    for (int j = 0; j < Nn; j += 4) {
        z0 += fexp2(lrelu01(tv + sh[j + 0]) - m2);
        z1 += fexp2(lrelu01(tv + sh[j + 1]) - m2);
        z2 += fexp2(lrelu01(tv + sh[j + 2]) - m2);
        z3 += fexp2(lrelu01(tv + sh[j + 3]) - m2);
    }
    float z = (z0 + z1) + (z2 + z3);
    Cout[h * Nn + i] = -m2 - flog2((float)Hh * z);
}

// ---- Stage 6: fused attention matmul. Block = 16 rows x 128 cols of out; 8 waves.
// weight = exp2(lrelu(t2+s2) + C); 16x32 tile built in LDS in WMMA-A order
// (double-buffered, one barrier per WMMA), v_wmma_f32_16x16x32_f16 vs f16 x.
__global__ void __launch_bounds__(256) k_attn(const _Float16* __restrict__ xT,
                                              const float* __restrict__ s,
                                              const float* __restrict__ t,
                                              const float* __restrict__ Carr,
                                              float* __restrict__ out) {
    __shared__ __align__(32) _Float16 eA[2][512];      // double-buffered A fragments
    __shared__ float lt[Hh][16], lC[Hh][16];
    int tid = threadIdx.x;
    int i0 = blockIdx.x * 16;

    if (tid < 64)       { int h = tid >> 4,        M = tid & 15; lt[h][M] = t   [h * Nn + i0 + M]; }
    else if (tid < 128) { int h = (tid - 64) >> 4, M = tid & 15; lC[h][M] = Carr[h * Nn + i0 + M]; }
    __syncthreads();

    int w = tid >> 5, l = tid & 31;                    // wave, lane
    // E-tile element pair owned by this thread: row M, cols K0,K0+1
    int M  = tid >> 4;                                 // 0..15
    int K0 = (tid & 15) << 1;                          // 0,2,...,30
    // A-fragment slot (this kernel's contiguous k order, k = 16*(lane/16) + slot): lanes<16 hold K 0..15, lanes>=16 hold K 16..31
    int lane_e = M + ((K0 >> 4) << 4);
    int p      = K0 & 15;
    // B fragment source: lanes<16 -> K=0..15, lanes>=16 -> K=16..31; col = w*16 + l%16
    int col = w * 16 + (l & 15);
    int ks  = (l < 16) ? 0 : 16;
    const _Float16* bbase = xT + (size_t)col * Nn + ks;

    // hoist per-(head,row) constants out of the j-loop (regs, not LDS reloads)
    float tva[Hh], Ca[Hh];
    #pragma unroll
    for (int h = 0; h < Hh; ++h) { tva[h] = lt[h][M]; Ca[h] = lC[h][M]; }

    v8f acc = {};
    int par = 0;
    for (int j0 = 0; j0 < Nn; j0 += 32) {
        v16h bfrag = *(const v16h*)(bbase + j0);       // reused across heads
        #pragma unroll
        for (int h = 0; h < Hh; ++h) {
            float s0 = s[h * Nn + j0 + K0];
            float s1 = s[h * Nn + j0 + K0 + 1];
            float v0 = tva[h] + s0; v0 = v0 > 0.f ? v0 : 0.01f * v0;
            float v1 = tva[h] + s1; v1 = v1 > 0.f ? v1 : 0.01f * v1;
            float e0 = fexp2(v0 + Ca[h]);              // weight incl. 1/(H*Z)
            float e1 = fexp2(v1 + Ca[h]);
            _Float16* buf = &eA[par][0];
            buf[lane_e * 16 + p]     = (_Float16)e0;
            buf[lane_e * 16 + p + 1] = (_Float16)e1;
            __syncthreads();                           // single barrier per WMMA
            v16h afrag = *(const v16h*)&buf[l * 16];
            acc = __builtin_amdgcn_wmma_f32_16x16x32_f16(
                      false, afrag, false, bfrag, (short)0, acc, false, false);
            par ^= 1;
        }
    }
    // C layout: VGPR r -> row r (lanes 0-15) / r+8 (lanes 16-31), col = lane%16
    int ncol = w * 16 + (l & 15);
    int rb   = (l < 16) ? 0 : 8;
    #pragma unroll
    for (int r = 0; r < 8; ++r)
        out[(size_t)(i0 + r + rb) * Dd + ncol] = acc[r];
}

extern "C" void kernel_launch(void* const* d_in, const int* in_sizes, int n_in,
                              void* d_out, int out_size, void* d_ws, size_t ws_size,
                              hipStream_t stream) {
    (void)in_sizes; (void)n_in; (void)out_size; (void)ws_size;
    const float* x = (const float*)d_in[0];
    const float* W = (const float*)d_in[1];
    const float* b = (const float*)d_in[2];
    const float* a = (const float*)d_in[3];
    float* out = (float*)d_out;
    char* ws = (char*)d_ws;

    _Float16* xT   = (_Float16*)(ws + OFF_XT);
    float*    u    = (float*)(ws + OFF_U);
    float*    cvec = (float*)(ws + OFF_C);
    float*    s    = (float*)(ws + OFF_S);
    float*    t    = (float*)(ws + OFF_T);
    float*    smax = (float*)(ws + OFF_SMAX);
    float*    Carr = (float*)(ws + OFF_CC);

    k_castT<<<(Nn * Dd) / 256, 256, 0, stream>>>(x, xT);
    k_prep <<<1, 256, 0, stream>>>(W, b, a, u, cvec);
    k_st   <<<Nn / 256, 256, 0, stream>>>(x, u, cvec, s, t);
    k_smax <<<1, 256, 0, stream>>>(s, smax);
    k_zc   <<<(Nn * Hh) / 256, 256, 0, stream>>>(s, t, smax, Carr);
    k_attn <<<Nn / 16, 256, 0, stream>>>(xT, s, t, Carr, out);
}
